// MOTNet_5179730559747
// MI455X (gfx1250) — matrix operand order repaired and checked by reading, not compiled, not run on hardware
//
#include <hip/hip_runtime.h>
#include <stdint.h>

typedef _Float16 f16;
typedef _Float16 v16h __attribute__((ext_vector_type(16)));
typedef _Float16 v4h  __attribute__((ext_vector_type(4)));
typedef float    v8f  __attribute__((ext_vector_type(8)));

#define GB   128            // graphs
#define NPG  512            // nodes per graph
#define DF   128            // feature dim
#define NNODE (GB*NPG)      // 65536
#define NEDGE (GB*NPG*8)    // 524288
#define KP1  410
#define KP2  328

// ---------------------------------------------------------------- utilities
__global__ void k_zero_f32(float* __restrict__ p, int n) {
    int i = blockIdx.x * blockDim.x + threadIdx.x;
    if (i < n) p[i] = 0.0f;
}

__global__ void k_cvt_f16(f16* __restrict__ dst, const float* __restrict__ src, int n4) {
    int i = blockIdx.x * blockDim.x + threadIdx.x;
    if (i < n4) {
        float4 v = ((const float4*)src)[i];
        v4h h;
        h[0] = (f16)v.x; h[1] = (f16)v.y; h[2] = (f16)v.z; h[3] = (f16)v.w;
        ((v4h*)dst)[i] = h;
    }
}

// transpose 128x128 f32 weight -> [n][k] f16 (so B-fragment K-pairs are contiguous)
__global__ void k_wt(f16* __restrict__ dst, const float* __restrict__ w) {
    int n = blockIdx.x;            // 128 blocks
    int k = threadIdx.x;           // 128 threads, coalesced writes
    dst[n * DF + k] = (f16)w[k * DF + n];
}

// ------------------------------------------------------------- aggregation
// AGG[dst] += feat[src] over all edges (sum aggregation). 1 wave per edge,
// each lane handles a float4 (32*4 = 128 feats).
__global__ void __launch_bounds__(256) k_agg(float* __restrict__ agg,
                                             const float* __restrict__ feat,
                                             const int* __restrict__ srcI,
                                             const int* __restrict__ dstI) {
    int e    = blockIdx.x * 8 + (threadIdx.x >> 5);
    int lane = threadIdx.x & 31;
    int s = srcI[e], d = dstI[e];
    float4 v = ((const float4*)(feat + (size_t)s * DF))[lane];
    float* dp = agg + (size_t)d * DF + lane * 4;
    atomicAdd(dp + 0, v.x);
    atomicAdd(dp + 1, v.y);
    atomicAdd(dp + 2, v.z);
    atomicAdd(dp + 3, v.w);
}

// ------------------------------------------------------------------- GEMM
// H = relu( Ax @ WrT^T + Aagg @ WnT^T + bias ), all [65536 x 128].
// One wave -> one 16x16 tile. grid.x = 4096 row tiles, 8 waves cover N=128.
__global__ void __launch_bounds__(256) k_gemm(float* __restrict__ Hout,
                                              const f16* __restrict__ Ax,
                                              const f16* __restrict__ Aagg,
                                              const f16* __restrict__ WrT,   // [n][k] f16
                                              const f16* __restrict__ WnT,   // [n][k] f16
                                              const float* __restrict__ bias) {
    const int lane = threadIdx.x & 31;
    const int wave = threadIdx.x >> 5;
    const int row0 = blockIdx.x * 16;
    const int col0 = wave * 16;
    const int m    = lane & 15;      // A row within tile
    const int nn   = lane & 15;      // B/C col within tile
    const int half = lane >> 4;

    const f16* arow = Ax   + (size_t)(row0 + m) * DF;
    const f16* grow = Aagg + (size_t)(row0 + m) * DF;
    const f16* wr   = WrT  + (size_t)(col0 + nn) * DF;
    const f16* wn   = WnT  + (size_t)(col0 + nn) * DF;

    v8f acc = {0.f, 0.f, 0.f, 0.f, 0.f, 0.f, 0.f, 0.f};

#pragma unroll
    for (int kk = 0; kk < 4; ++kk) {
        const int ko = kk * 32;
        v16h a, g, br, bn;
        uint32_t* au  = (uint32_t*)&a;
        uint32_t* gu  = (uint32_t*)&g;
        uint32_t* bru = (uint32_t*)&br;
        uint32_t* bnu = (uint32_t*)&bn;
#pragma unroll
        for (int v = 0; v < 8; ++v) {
            // A 16x32 — lane-half selects K range 0-15/16-31, VGPR v holds K=2v,2v+1 (matches B)
            int ka = ko + half * 16 + 2 * v;
            // B 32x16 — lane-half selects K range 0-15/16-31, VGPR v holds K=2v,2v+1
            int kb = ko + half * 16 + 2 * v;
            au[v]  = *(const uint32_t*)(arow + ka);
            gu[v]  = *(const uint32_t*)(grow + ka);
            bru[v] = *(const uint32_t*)(wr + kb);
            bnu[v] = *(const uint32_t*)(wn + kb);
        }
        acc = __builtin_amdgcn_wmma_f32_16x16x32_f16(false, a, false, br, (short)0, acc, false, false);
        acc = __builtin_amdgcn_wmma_f32_16x16x32_f16(false, g, false, bn, (short)0, acc, false, false);
    }

    const float bv = bias[col0 + nn];
#pragma unroll
    for (int v = 0; v < 8; ++v) {
        float val = acc[v] + bv;
        val = val > 0.f ? val : 0.f;                      // ReLU
        Hout[(size_t)(row0 + v + half * 8) * DF + col0 + nn] = val;
    }
}

// ------------------------------------------------------------------ scores
// S[i] = dot(H[i,:], p) / ||p||  — one wave per node, float4 per lane.
__global__ void __launch_bounds__(128) k_score(float* __restrict__ S,
                                               const float* __restrict__ H,
                                               const float* __restrict__ p) {
    int lane = threadIdx.x & 31;
    int node = blockIdx.x * 4 + (threadIdx.x >> 5);
    float4 h = ((const float4*)(H + (size_t)node * DF))[lane];
    float4 q = ((const float4*)p)[lane];
    float s  = h.x * q.x + h.y * q.y + h.z * q.z + h.w * q.w;
    float pn = q.x * q.x + q.y * q.y + q.z * q.z + q.w * q.w;
#pragma unroll
    for (int off = 16; off > 0; off >>= 1) {
        s  += __shfl_xor(s, off, 32);
        pn += __shfl_xor(pn, off, 32);
    }
    if (lane == 0) S[node] = s / sqrtf(pn);
}

// ------------------------------------------------------------------- top-k
// per-graph bitonic sort of 512 (score,idx) pairs in LDS, descending;
// rank < k  ->  mask = 1. prevMask restricts candidates (NULL = all valid).
__global__ void __launch_bounds__(512) k_topk(int* __restrict__ mask,
                                              const float* __restrict__ S,
                                              const int* __restrict__ prevMask,
                                              int k) {
    __shared__ float ks[NPG];
    __shared__ short vs[NPG];
    const int g = blockIdx.x, t = threadIdx.x;
    float s = S[g * NPG + t];
    if (prevMask && !prevMask[g * NPG + t]) s = -__builtin_inff();
    ks[t] = s; vs[t] = (short)t;
    __syncthreads();
    for (int size = 2; size <= NPG; size <<= 1) {
        for (int stride = size >> 1; stride > 0; stride >>= 1) {
            int q = t ^ stride;
            if (q > t) {
                bool desc = (t & size) == 0;
                float a = ks[t], b = ks[q];
                if (desc ? (a < b) : (a > b)) {
                    ks[t] = b; ks[q] = a;
                    short tv = vs[t]; vs[t] = vs[q]; vs[q] = tv;
                }
            }
            __syncthreads();
        }
    }
    mask[g * NPG + vs[t]] = (t < k) ? 1 : 0;
}

// ------------------------------------------------------------------- gate
__global__ void __launch_bounds__(256) k_gate(float* __restrict__ H,
                                              const float* __restrict__ S,
                                              const int* __restrict__ mask) {
    int i = blockIdx.x * blockDim.x + threadIdx.x;   // float4 granules
    int node = i >> 5;                               // 32 granules per node
    float gsc = tanhf(S[node]) * (mask[node] ? 1.f : 0.f);
    float4 v = ((float4*)H)[i];
    v.x *= gsc; v.y *= gsc; v.z *= gsc; v.w *= gsc;
    ((float4*)H)[i] = v;
}

// ----------------------------------------------------------------- readout
// Z[g, 0:128]   (+)= max over kept nodes of gated h
// Z[g, 128:256] (+)= sum(gated h)/k   (dropped nodes are exactly zero)
__global__ void __launch_bounds__(128) k_readout(float* __restrict__ Z,
                                                 const float* __restrict__ H,
                                                 const int* __restrict__ mask,
                                                 float invk, int accum) {
    const int g = blockIdx.x, f = threadIdx.x;
    const float* hg = H + (size_t)g * NPG * DF;
    const int*   mg = mask + g * NPG;
    float gmax = -__builtin_inff();
    float sum = 0.f;
    for (int n = 0; n < NPG; ++n) {
        float v = hg[(size_t)n * DF + f];
        if (mg[n]) gmax = fmaxf(gmax, v);
        sum += v;
    }
    float mean = sum * invk;
    if (accum) {
        Z[g * 256 + f]       += gmax;
        Z[g * 256 + 128 + f] += mean;
    } else {
        Z[g * 256 + f]       = gmax;
        Z[g * 256 + 128 + f] = mean;
    }
}

// -------------------------------------------------------------------- MLP
__global__ void __launch_bounds__(128) k_mlp(float* __restrict__ out,
                                             const float* __restrict__ Z,
                                             const float* __restrict__ lw1, const float* __restrict__ lb1,
                                             const float* __restrict__ lw2, const float* __restrict__ lb2,
                                             const float* __restrict__ lw3, const float* __restrict__ lb3) {
    __shared__ float zin[256], a1[128], a2[64];
    const int g = blockIdx.x, t = threadIdx.x;
    zin[t]       = Z[g * 256 + t];
    zin[t + 128] = Z[g * 256 + 128 + t];
    __syncthreads();
    float acc = lb1[t];
    for (int j = 0; j < 256; ++j) acc += zin[j] * lw1[j * 128 + t];
    a1[t] = fmaxf(acc, 0.f);
    __syncthreads();
    if (t < 64) {
        float acc2 = lb2[t];
        for (int j = 0; j < 128; ++j) acc2 += a1[j] * lw2[j * 64 + t];
        a2[t] = fmaxf(acc2, 0.f);
    }
    __syncthreads();
    if (t == 0) {
        float acc3 = lb3[0];
        for (int j = 0; j < 64; ++j) acc3 += a2[j] * lw3[j];
        out[g] = 1.f / (1.f + expf(-acc3));
    }
}

// ------------------------------------------------------------------ driver
extern "C" void kernel_launch(void* const* d_in, const int* in_sizes, int n_in,
                              void* d_out, int out_size, void* d_ws, size_t ws_size,
                              hipStream_t stream) {
    const float* x   = (const float*)d_in[0];
    const int*   ei  = (const int*)  d_in[1];
    const float* W1r = (const float*)d_in[2];
    const float* W1n = (const float*)d_in[3];
    const float* b1  = (const float*)d_in[4];
    const float* p1  = (const float*)d_in[5];
    const float* W2r = (const float*)d_in[6];
    const float* W2n = (const float*)d_in[7];
    const float* b2  = (const float*)d_in[8];
    const float* p2  = (const float*)d_in[9];
    const float* lw1 = (const float*)d_in[10];
    const float* lb1 = (const float*)d_in[11];
    const float* lw2 = (const float*)d_in[12];
    const float* lb2 = (const float*)d_in[13];
    const float* lw3 = (const float*)d_in[14];
    const float* lb3 = (const float*)d_in[15];

    const int* srcI = ei;            // edge_index[0]
    const int* dstI = ei + NEDGE;    // edge_index[1]

    const size_t FEATB = (size_t)NNODE * DF;   // 8388608 elements
    char* ws = (char*)d_ws;
    float* AGG  = (float*)ws;            ws += FEATB * sizeof(float);   // 32 MB
    float* H    = (float*)ws;            ws += FEATB * sizeof(float);   // 32 MB
    f16*   XH   = (f16*)ws;              ws += FEATB * sizeof(f16);     // 16 MB
    f16*   AGGH = (f16*)ws;              ws += FEATB * sizeof(f16);     // 16 MB
    f16*   WT   = (f16*)ws;              ws += 2 * DF * DF * sizeof(f16);
    float* S    = (float*)ws;            ws += NNODE * sizeof(float);
    int*   M1   = (int*)ws;              ws += NNODE * sizeof(int);
    int*   M2   = (int*)ws;              ws += NNODE * sizeof(int);
    float* Z    = (float*)ws;            ws += GB * 256 * sizeof(float);

    const int Z256 = 256;
    const int nFeatBlk = (int)(FEATB / Z256);          // 32768
    const int nCvtBlk  = (int)(FEATB / 4 / Z256);      // 8192

    for (int round = 0; round < 2; ++round) {
        const float* srcFeat  = (round == 0) ? x   : H;   // round 2 reads gated h1
        const float* Wr       = (round == 0) ? W1r : W2r;
        const float* Wn       = (round == 0) ? W1n : W2n;
        const float* bb       = (round == 0) ? b1  : b2;
        const float* pp       = (round == 0) ? p1  : p2;
        const int    kkeep    = (round == 0) ? KP1 : KP2;
        const int*   prevMask = (round == 0) ? (const int*)nullptr : M1;
        int*         outMask  = (round == 0) ? M1 : M2;

        // snapshot input before GEMM overwrites H in place
        k_cvt_f16 <<<nCvtBlk, Z256, 0, stream>>>(XH, srcFeat, (int)(FEATB / 4));
        k_zero_f32<<<nFeatBlk, Z256, 0, stream>>>(AGG, (int)FEATB);
        k_agg     <<<NEDGE / 8, 256, 0, stream>>>(AGG, srcFeat, srcI, dstI);
        k_cvt_f16 <<<nCvtBlk, Z256, 0, stream>>>(AGGH, AGG, (int)(FEATB / 4));
        k_wt      <<<DF, DF, 0, stream>>>(WT, Wr);
        k_wt      <<<DF, DF, 0, stream>>>(WT + DF * DF, Wn);
        k_gemm    <<<NNODE / 16, 256, 0, stream>>>(H, XH, AGGH, WT, WT + DF * DF, bb);
        k_score   <<<NNODE / 4, 128, 0, stream>>>(S, H, pp);
        k_topk    <<<GB, NPG, 0, stream>>>(outMask, S, prevMask, kkeep);
        k_gate    <<<nCvtBlk, Z256, 0, stream>>>(H, S, outMask);
        k_readout <<<GB, DF, 0, stream>>>(Z, H, outMask, 1.0f / (float)kkeep, round);
    }

    k_mlp<<<GB, 128, 0, stream>>>((float*)d_out, Z, lw1, lb1, lw2, lb2, lw3, lb3);
}
